// CrossAttentionBlock_25159918420593
// MI455X (gfx1250) — matrix operand order repaired and checked by reading, not compiled, not run on hardware
//
#include <hip/hip_runtime.h>
#include <hip/hip_bf16.h>
#include <cstdint>

// ---------------------------------------------------------------------------
// CrossAttentionBlock on MI455X (gfx1250): all GEMMs via v_wmma_f32_16x16x32_bf16
// D = 128 (4 WMMA K-chunks), B=64, LMAX=256, AMAX=1024
// Fragments loaded from LDS as 16-byte ds_load_b128 chunks; B-operands staged
// transposed; epilogues specialized at compile time; fast-rcp sigmoid.
// ---------------------------------------------------------------------------

#define HDIM 128
#define NSEG 64
#define LMAX 256
#define AMAX 1024

typedef __bf16 bf16;
typedef __attribute__((ext_vector_type(16))) __bf16 v16bf;
typedef __attribute__((ext_vector_type(8)))  __bf16 v8bf;
typedef __attribute__((ext_vector_type(4)))  __bf16 v4bf;
typedef __attribute__((ext_vector_type(2)))  __bf16 v2bf;
typedef __attribute__((ext_vector_type(8)))  float  v8f;

__device__ __forceinline__ bf16 f2bf(float x) { return (bf16)x; }
__device__ __forceinline__ float leaky(float x) { return fmaxf(x, 0.1f * x); }

// build a v16bf WMMA operand from two contiguous 16-byte LDS chunks
__device__ __forceinline__ v16bf frag16(const bf16* p_lo, const bf16* p_hi) {
    v8bf lo = *(const v8bf*)p_lo;
    v8bf hi = *(const v8bf*)p_hi;
    v16bf r;
    #pragma unroll
    for (int j = 0; j < 8; ++j) { r[j] = lo[j]; r[8 + j] = hi[j]; }
    return r;
}

enum {
    EP_LEAKY = 1,   // y = leaky(y)
    EP_FINAL = 2,   // y = y * (1/len_other[seg]) + resid;  y = 0 for padded rows
    ST_F32   = 4,   // store float32 to Yf
    ST_BF16  = 8,   // store bf16 to Yb
    X_BF16   = 16,  // X input is bf16 (else f32)
};

// ---------------------------------------------------------------------------
// Generic  Y[M,128] = epilogue( X[M,128] @ W[128,128] + bias )
// block = 256 threads (8 waves), 64 rows per block, full 128 cols
// wave w: row-strip (w&3)*16, column half (w>>2)*64 (4 n-tiles of 16)
// W staged TRANSPOSED in LDS so B-fragments are contiguous.
// ---------------------------------------------------------------------------
template <int FLAGS>
__global__ __launch_bounds__(256)
void gemm128_wmma_kernel(const void* __restrict__ Xv,
                         const float* __restrict__ W,
                         const float* __restrict__ bias,
                         float* __restrict__ Yf,
                         bf16*  __restrict__ Yb,
                         const float* __restrict__ resid,
                         const int* __restrict__ len_self,
                         const int* __restrict__ len_other,
                         int nshift)
{
    __shared__ __align__(16) bf16 lx[64 * HDIM];      // 16 KB  X tile (row-major)
    __shared__ __align__(16) bf16 lwT[HDIM * HDIM];   // 32 KB  W transposed [n][k]
    __shared__ float lb[HDIM];                        // bias

    const int tid  = threadIdx.x;
    const int row0 = blockIdx.x * 64;

    // stage W -> bf16, transposed (lwT[n*128 + k] = W[k*128 + n])
    {
        const float4* W4 = (const float4*)W;
        for (int i = tid; i < HDIM * HDIM / 4; i += 256) {
            const int k = (i * 4) >> 7;
            const int n = (i * 4) & 127;
            float4 w = W4[i];
            lwT[(n + 0) * HDIM + k] = f2bf(w.x);
            lwT[(n + 1) * HDIM + k] = f2bf(w.y);
            lwT[(n + 2) * HDIM + k] = f2bf(w.z);
            lwT[(n + 3) * HDIM + k] = f2bf(w.w);
        }
    }
    if (tid < HDIM) lb[tid] = bias[tid];
    // stage X tile (row-major, vectorized)
    if constexpr (FLAGS & X_BF16) {
        const v8bf* X8 = (const v8bf*)((const bf16*)Xv + (size_t)row0 * HDIM);
        v8bf* lx8 = (v8bf*)lx;
        for (int i = tid; i < 64 * HDIM / 8; i += 256) lx8[i] = X8[i];
    } else {
        const float4* X4 = (const float4*)((const float*)Xv + (size_t)row0 * HDIM);
        v4bf* lx4 = (v4bf*)lx;
        for (int i = tid; i < 64 * HDIM / 4; i += 256) {
            float4 x = X4[i];
            v4bf b = { f2bf(x.x), f2bf(x.y), f2bf(x.z), f2bf(x.w) };
            lx4[i] = b;
        }
    }
    __syncthreads();

    const int wave  = tid >> 5;
    const int lane  = tid & 31;
    const int strip = wave & 3;       // 16-row strip within the 64-row tile
    const int nhalf = wave >> 2;      // 0 -> cols 0..63, 1 -> cols 64..127
    const int m     = lane & 15;
    const int hi    = lane >> 4;      // lane half
    const int lrow0 = strip * 16 + hi * 8;   // first of this lane's 8 output rows

    // block-uniform epilogue scalars (whole 64-row tile is inside one segment)
    float invn = 0.f;
    int rem = 0x7fffffff;             // rows (from lrow0) still inside the mask
    if constexpr (FLAGS & EP_FINAL) {
        const int seg = row0 >> nshift;
        invn = 1.0f / (float)len_other[seg];
        rem  = len_self[seg] - (row0 & ((1 << nshift) - 1)) - lrow0;
    }

    // A fragments (16x32 bf16), 4 K-chunks covering K=0..127
    v16bf afrag[4];
    {
        const bf16* arow = lx + (strip * 16 + m) * HDIM + hi * 8;
        #pragma unroll
        for (int c = 0; c < 4; ++c)
            afrag[c] = frag16(arow + c * 32, arow + c * 32 + 16);
    }

    #pragma unroll
    for (int t = 0; t < 4; ++t) {
        const int col = nhalf * 64 + t * 16 + m;
        const float bv = lb[col];
        v8f acc;
        #pragma unroll
        for (int r = 0; r < 8; ++r) acc[r] = bv;

        const bf16* bcol = lwT + col * HDIM + hi * 8;
        #pragma unroll
        for (int c = 0; c < 4; ++c) {
            v16bf bfrag = frag16(bcol + c * 32, bcol + c * 32 + 16);
            acc = __builtin_amdgcn_wmma_f32_16x16x32_bf16(
                false, afrag[c], false, bfrag, (short)0, acc, false, false);
        }

        // hoisted per-lane base pointers: per-r accesses use immediate offsets
        const size_t base = (size_t)(row0 + lrow0) * HDIM + col;
        const float* rp = (FLAGS & EP_FINAL) ? resid + base : nullptr;
        float* yf = (FLAGS & ST_F32) ? Yf + base : nullptr;
        bf16*  yb = (FLAGS & ST_BF16) ? Yb + base : nullptr;

        #pragma unroll
        for (int r = 0; r < 8; ++r) {
            float y = acc[r];
            if constexpr (FLAGS & EP_LEAKY) y = leaky(y);
            if constexpr (FLAGS & EP_FINAL) {
                y = y * invn + rp[r * HDIM];
                y = (r < rem) ? y : 0.f;
            }
            if constexpr (FLAGS & ST_F32)  yf[r * HDIM] = y;
            if constexpr (FLAGS & ST_BF16) yb[r * HDIM] = f2bf(y);
        }
    }
}

// ---------------------------------------------------------------------------
// Per-segment:  Msg = (sigmoid(Q @ K^T) * mask_keys) @ V
// grid = (NQ/64, NSEG); block = 256 threads = 8 waves; 64 query rows / block.
// K chunk staged row-major (contiguous B-frags for Q@K^T);
// V chunk staged TRANSPOSED (contiguous B-frags for S@V), packed b32 stores.
// ---------------------------------------------------------------------------
__global__ __launch_bounds__(256)
void sigmoid_attn_wmma_kernel(const bf16* __restrict__ Q,
                              const bf16* __restrict__ K,
                              const bf16* __restrict__ V,
                              bf16* __restrict__ Msg,
                              const int* __restrict__ len_k,
                              int NQ, int NK)
{
    __shared__ __align__(16) bf16 lq[64 * HDIM];    // 16 KB  queries (row-major)
    __shared__ __align__(16) bf16 lk[64 * HDIM];    // 16 KB  keys (row-major)
    __shared__ __align__(16) bf16 lvT[HDIM * 64];   // 16 KB  values transposed [col][key]
    __shared__ __align__(16) bf16 ls[64 * 64];      //  8 KB  scores (bf16)

    const int tid = threadIdx.x;
    const int seg = blockIdx.y;
    const int q0  = blockIdx.x * 64;
    const bf16* Qs = Q + (size_t)seg * NQ * HDIM;
    const bf16* Ks = K + (size_t)seg * NK * HDIM;
    const bf16* Vs = V + (size_t)seg * NK * HDIM;
    const int klen = len_k[seg];

    {
        const v8bf* Q8 = (const v8bf*)(Qs + (size_t)q0 * HDIM);
        v8bf* lq8 = (v8bf*)lq;
        for (int i = tid; i < 64 * HDIM / 8; i += 256) lq8[i] = Q8[i];
    }
    __syncthreads();

    const int wave  = tid >> 5;
    const int lane  = tid & 31;
    const int strip = wave & 3;
    const int nh    = wave >> 2;
    const int m     = lane & 15;
    const int hi    = lane >> 4;
    const int lrow0 = strip * 16 + hi * 8;

    // Q A-fragments for this wave's 16-row strip (resident for whole kernel)
    v16bf qf[4];
    {
        const bf16* qrow = lq + (strip * 16 + m) * HDIM + hi * 8;
        #pragma unroll
        for (int c = 0; c < 4; ++c)
            qf[c] = frag16(qrow + c * 32, qrow + c * 32 + 16);
    }

    // message accumulators: 4 col-tiles of 16 (cols nh*64 .. nh*64+63)
    v8f macc[4];
    #pragma unroll
    for (int t = 0; t < 4; ++t)
        #pragma unroll
        for (int r = 0; r < 8; ++r) macc[t][r] = 0.f;

    for (int c0 = 0; c0 < NK; c0 += 64) {
        // prefetch next K/V chunk into cache (global_prefetch_b8), one
        // cacheline (64 B = 32 bf16) per thread covers the 16 KB chunk
        if (c0 + 64 < NK) {
            __builtin_prefetch(Ks + (size_t)(c0 + 64) * HDIM + tid * 32, 0, 3);
            __builtin_prefetch(Vs + (size_t)(c0 + 64) * HDIM + tid * 32, 0, 3);
        }
        __syncthreads();   // protect lk/lvT/ls reuse from previous iteration
        {
            // keys: straight vector copy
            const v8bf* K8 = (const v8bf*)(Ks + (size_t)c0 * HDIM);
            v8bf* lk8 = (v8bf*)lk;
            for (int i = tid; i < 64 * HDIM / 8; i += 256) lk8[i] = K8[i];
            // values: transpose, two keys at a time -> packed b32 LDS stores
            const v8bf* V8 = (const v8bf*)(Vs + (size_t)c0 * HDIM);
            for (int i = tid; i < 64 * HDIM / 16; i += 256) {
                const int r  = (i >> 4) * 2;     // even key index
                const int cb = (i & 15) * 8;     // feature base
                v8bf v0 = V8[(r * HDIM + cb) >> 3];
                v8bf v1 = V8[((r + 1) * HDIM + cb) >> 3];
                #pragma unroll
                for (int j = 0; j < 8; ++j) {
                    v2bf p = { v0[j], v1[j] };
                    *(v2bf*)&lvT[(cb + j) * 64 + r] = p;
                }
            }
        }
        __syncthreads();

        // ---- S tile: rows = strip, col-tiles ct in {2*nh, 2*nh+1} ----
        #pragma unroll
        for (int t = 0; t < 2; ++t) {
            const int ct = nh * 2 + t;
            v8f s;
            #pragma unroll
            for (int r = 0; r < 8; ++r) s[r] = 0.f;
            const bf16* krow = lk + (ct * 16 + m) * HDIM + hi * 8;
            #pragma unroll
            for (int c = 0; c < 4; ++c) {
                v16bf bfrag = frag16(krow + c * 32, krow + c * 32 + 16);
                s = __builtin_amdgcn_wmma_f32_16x16x32_bf16(
                    false, qf[c], false, bfrag, (short)0, s, false, false);
            }
            // sigmoid via fast v_rcp + key-mask, write bf16 scores to LDS
            const int kglob = c0 + ct * 16 + m;
            const float msk = (kglob < klen) ? 1.f : 0.f;
            bf16* sp = ls + lrow0 * 64 + ct * 16 + m;
            #pragma unroll
            for (int r = 0; r < 8; ++r) {
                const float sig = __builtin_amdgcn_rcpf(1.f + __expf(-s[r])) * msk;
                sp[r * 64] = f2bf(sig);
            }
        }
        __syncthreads();

        // ---- msg += S @ V_chunk  (K-dim = 64 keys = 2 WMMA chunks) ----
        v16bf sf[2];
        {
            const bf16* srow = ls + (strip * 16 + m) * 64 + hi * 8;
            #pragma unroll
            for (int c2 = 0; c2 < 2; ++c2)
                sf[c2] = frag16(srow + c2 * 32, srow + c2 * 32 + 16);
        }
        #pragma unroll
        for (int t = 0; t < 4; ++t) {
            const int col = nh * 64 + t * 16 + m;
            const bf16* vcol = lvT + col * 64 + hi * 8;
            #pragma unroll
            for (int c2 = 0; c2 < 2; ++c2) {
                v16bf bfrag = frag16(vcol + c2 * 32, vcol + c2 * 32 + 16);
                macc[t] = __builtin_amdgcn_wmma_f32_16x16x32_bf16(
                    false, sf[c2], false, bfrag, (short)0, macc[t], false, false);
            }
        }
    }

    // store message (bf16, consumed by the MLP GEMM)
    #pragma unroll
    for (int t = 0; t < 4; ++t) {
        const int col = nh * 64 + t * 16 + m;
        bf16* mp = Msg + ((size_t)seg * NQ + q0 + lrow0) * HDIM + col;
        #pragma unroll
        for (int r = 0; r < 8; ++r)
            mp[r * HDIM] = f2bf(macc[t][r]);
    }
}

// ---------------------------------------------------------------------------
extern "C" void kernel_launch(void* const* d_in, const int* in_sizes, int n_in,
                              void* d_out, int out_size, void* d_ws, size_t ws_size,
                              hipStream_t stream)
{
    const float* lig = (const float*)d_in[0];
    const float* aa  = (const float*)d_in[1];
    const int* lig_len = (const int*)d_in[2];
    const int* aa_len  = (const int*)d_in[3];
    const float* Wq = (const float*)d_in[4],  *bq = (const float*)d_in[5];
    const float* Wk = (const float*)d_in[6],  *bk = (const float*)d_in[7];
    const float* Wv = (const float*)d_in[8],  *bv = (const float*)d_in[9];
    const float* Wr1 = (const float*)d_in[10], *br1 = (const float*)d_in[11];
    const float* Wr2 = (const float*)d_in[12], *br2 = (const float*)d_in[13];
    const float* Wl1 = (const float*)d_in[14], *bl1 = (const float*)d_in[15];
    const float* Wl2 = (const float*)d_in[16], *bl2 = (const float*)d_in[17];

    const size_t SZ_A = (size_t)NSEG * AMAX * HDIM;  // 8,388,608
    const size_t SZ_L = (size_t)NSEG * LMAX * HDIM;  // 2,097,152

    // bf16 workspace carve-out (phase-2 reuses phase-1 regions)
    bf16* qb   = (bf16*)d_ws;        // [B*AMAX,128]
    bf16* kb   = qb + SZ_A;          // [B*LMAX,128]
    bf16* vb   = kb + SZ_L;          // [B*LMAX,128]
    bf16* msgb = vb + SZ_L;          // [B*AMAX,128]
    bf16* h1b  = msgb + SZ_A;        // [B*AMAX,128]
    bf16* q2b  = kb;                 // reuse
    bf16* k2b  = qb;                 // reuse
    bf16* v2b  = msgb;               // reuse
    bf16* msg2b = vb;                // reuse
    bf16* h2b  = h1b;                // reuse

    float* lig_out = (float*)d_out;            // [B,LMAX,D]
    float* aa_out  = lig_out + SZ_L;           // [B,AMAX,D]

    const dim3 blk(256);
    const int GA = (NSEG * AMAX) / 64;   // 1024 blocks for M = 65536
    const int GL = (NSEG * LMAX) / 64;   // 256  blocks for M = 16384
    const int ASHIFT = 10;               // log2(AMAX)
    const int LSHIFT = 8;                // log2(LMAX)

    constexpr int F_PROJ  = ST_BF16;                                   // f32 X -> bf16 proj
    constexpr int F_HID   = ST_BF16 | EP_LEAKY | X_BF16;               // MLP hidden
    constexpr int F_FINAL = ST_F32 | EP_LEAKY | EP_FINAL | X_BF16;     // MLP out + residual

    // ---------- phase 1: aa update ----------
    gemm128_wmma_kernel<F_PROJ><<<GA, blk, 0, stream>>>(aa,  Wq, bq, nullptr, qb, nullptr, nullptr, nullptr, 0);
    gemm128_wmma_kernel<F_PROJ><<<GL, blk, 0, stream>>>(lig, Wk, bk, nullptr, kb, nullptr, nullptr, nullptr, 0);
    gemm128_wmma_kernel<F_PROJ><<<GL, blk, 0, stream>>>(lig, Wv, bv, nullptr, vb, nullptr, nullptr, nullptr, 0);

    sigmoid_attn_wmma_kernel<<<dim3(AMAX / 64, NSEG), blk, 0, stream>>>(
        qb, kb, vb, msgb, lig_len, AMAX, LMAX);

    gemm128_wmma_kernel<F_HID><<<GA, blk, 0, stream>>>(msgb, Wr1, br1, nullptr, h1b, nullptr, nullptr, nullptr, 0);
    gemm128_wmma_kernel<F_FINAL><<<GA, blk, 0, stream>>>(h1b, Wr2, br2, aa_out, nullptr, aa, aa_len, lig_len, ASHIFT);

    // ---------- phase 2: ligand update (consumes aa_out) ----------
    gemm128_wmma_kernel<F_PROJ><<<GL, blk, 0, stream>>>(lig,    Wq, bq, nullptr, q2b, nullptr, nullptr, nullptr, 0);
    gemm128_wmma_kernel<F_PROJ><<<GA, blk, 0, stream>>>(aa_out, Wk, bk, nullptr, k2b, nullptr, nullptr, nullptr, 0);
    gemm128_wmma_kernel<F_PROJ><<<GA, blk, 0, stream>>>(aa_out, Wv, bv, nullptr, v2b, nullptr, nullptr, nullptr, 0);

    sigmoid_attn_wmma_kernel<<<dim3(LMAX / 64, NSEG), blk, 0, stream>>>(
        q2b, k2b, v2b, msg2b, aa_len, LMAX, AMAX);

    gemm128_wmma_kernel<F_HID><<<GL, blk, 0, stream>>>(msg2b, Wl1, bl1, nullptr, h2b, nullptr, nullptr, nullptr, 0);
    gemm128_wmma_kernel<F_FINAL><<<GL, blk, 0, stream>>>(h2b, Wl2, bl2, lig_out, nullptr, lig, lig_len, aa_len, LSHIFT);
}
